// TransformerSelfAttention_6975026888775
// MI455X (gfx1250) — matrix operand order repaired and checked by reading, not compiled, not run on hardware
//
#include <hip/hip_runtime.h>

// ---------------------------------------------------------------------------
// Transformer self-attention forward for MI455X (gfx1250, wave32, WMMA).
// Precision: bf16 inputs to v_wmma_f32_16x16x32_bf16, f32 accumulation.
// Attention K/Q tiles staged with GLOBAL_LOAD_ASYNC_TO_LDS_B128 (ASYNCcnt).
// ---------------------------------------------------------------------------

typedef __bf16 v16bf __attribute__((ext_vector_type(16)));
typedef float  v8f   __attribute__((ext_vector_type(8)));

#define B_  2
#define S_  2048
#define H_  1024
#define NH_ 16
#define DH_ 64
#define M_  (B_ * S_)        // 4096 tokens

union Frag16 { v16bf v; uint4 u[2]; };   // 32 bytes: one WMMA A/B operand per lane

// ---------------------------------------------------------------------------
// Async global->LDS copy of one 64-byte per-lane chunk (4 x b128).
// GVS mode: saddr = uniform 64-bit base, vaddr = per-lane 32-bit byte offset.
// Per lane: 16 bytes from saddr+vaddr+offset are written to LDS at vdst+offset.
// (IOFFSET applies to BOTH sides; our chunks are contiguous in both spaces.)
// ---------------------------------------------------------------------------
static __device__ __forceinline__
void async_copy64(unsigned lds_addr, const void* sbase, unsigned voff) {
  asm volatile(
      "global_load_async_to_lds_b128 %0, %1, %2 offset:0\n\t"
      "global_load_async_to_lds_b128 %0, %1, %2 offset:16\n\t"
      "global_load_async_to_lds_b128 %0, %1, %2 offset:32\n\t"
      "global_load_async_to_lds_b128 %0, %1, %2 offset:48"
      :: "v"(lds_addr), "v"(voff), "s"(sbase) : "memory");
}
static __device__ __forceinline__ void wait_async0() {
  asm volatile("s_wait_asynccnt 0x0" ::: "memory");
}
static __device__ __forceinline__ unsigned lds_off(const void* p) {
  // Generic pointers to LDS carry the byte offset in the low 32 bits.
  return (unsigned)(uintptr_t)p;
}

// ---------------------------------------------------------------------------
// f32 -> bf16 cast (vectorized 4-wide)
// ---------------------------------------------------------------------------
__global__ __launch_bounds__(256)
void cast_f32_bf16(const float* __restrict__ in, __bf16* __restrict__ out, int n4) {
  int i = blockIdx.x * blockDim.x + threadIdx.x;
  if (i >= n4) return;
  const float4 f = ((const float4*)in)[i];
  union { __bf16 h[4]; uint2 u; } p;
  p.h[0] = (__bf16)f.x; p.h[1] = (__bf16)f.y;
  p.h[2] = (__bf16)f.z; p.h[3] = (__bf16)f.w;
  ((uint2*)out)[i] = p.u;
}

// ---------------------------------------------------------------------------
// QKV projection GEMM: out[m,n] = sum_k X[m,k] * W[n,k] + bias[n]
// z = blockIdx.z selects Wq/Wk/Wv.  Q and K are multiplied by att_mask[n].
// Output layout: qkv[z][b][h][s][d] (head-major, bf16).
// Block tile 128x128, K-step 32, 8 waves of 32, each wave -> 32x64 outputs.
// (Kept on the VGPR-bounce path: the scheduler fully overlaps it with WMMA.)
// ---------------------------------------------------------------------------
#define BM  128
#define BN  128
#define BK  32
#define LDA 40   // padded LDS row stride (halves): 20 dwords, conflict-free

__global__ __launch_bounds__(256)
void qkv_gemm(const __bf16* __restrict__ X, const __bf16* __restrict__ Wall,
              const float* __restrict__ bq, const float* __restrict__ bk,
              const float* __restrict__ bv, const float* __restrict__ attm,
              __bf16* __restrict__ qkv)
{
  __shared__ __bf16 As[BM][LDA];
  __shared__ __bf16 Bs[BN][LDA];

  const int z    = blockIdx.z;
  const int bn   = blockIdx.x * BN;
  const int bm   = blockIdx.y * BM;
  const int tid  = threadIdx.x;
  const int lane = tid & 31, wid = tid >> 5;
  const int l16  = lane & 15, lh = lane >> 4;
  const int wm   = (wid >> 1) * 32;     // wave row offset in block tile
  const int wn   = (wid & 1) * 64;      // wave col offset in block tile

  const __bf16* Wp   = Wall + (size_t)z * (H_ * H_);
  const float*  bias = (z == 0) ? bq : (z == 1) ? bk : bv;

  const int trow = tid >> 1;
  const int tcol = (tid & 1) * 16;

  v8f c[2][4];
  for (int i = 0; i < 2; ++i)
    for (int j = 0; j < 4; ++j)
      for (int r = 0; r < 8; ++r)
        c[i][j][r] = 0.0f;

  for (int kk = 0; kk < H_; kk += BK) {
    __syncthreads();
    { // stage A tile (128x32): 2 threads per row, 32B each
      const uint4* g = (const uint4*)(X + (size_t)(bm + trow) * H_ + kk + tcol);
      *(uint4*)&As[trow][tcol]     = g[0];
      *(uint4*)&As[trow][tcol + 8] = g[1];
    }
    { // stage B tile = W rows (128x32)
      const uint4* g = (const uint4*)(Wp + (size_t)(bn + trow) * H_ + kk + tcol);
      *(uint4*)&Bs[trow][tcol]     = g[0];
      *(uint4*)&Bs[trow][tcol + 8] = g[1];
    }
    __syncthreads();

    Frag16 a[2], bfr[4];
    #pragma unroll
    for (int i = 0; i < 2; ++i) {  // A layout: lane=row, K split per lane-half
      const int m = wm + i * 16 + l16;
      a[i].u[0] = *(const uint4*)&As[m][lh * 8];
      a[i].u[1] = *(const uint4*)&As[m][16 + lh * 8];
    }
    #pragma unroll
    for (int j = 0; j < 4; ++j) {  // B layout: lane=col, K split per lane-half
      const int n = wn + j * 16 + l16;
      bfr[j].u[0] = *(const uint4*)&Bs[n][lh * 8];
      bfr[j].u[1] = *(const uint4*)&Bs[n][16 + lh * 8];
    }
    #pragma unroll
    for (int i = 0; i < 2; ++i)
      #pragma unroll
      for (int j = 0; j < 4; ++j)
        c[i][j] = __builtin_amdgcn_wmma_f32_16x16x32_bf16(
            false, a[i].v, false, bfr[j].v, (short)0, c[i][j], false, false);
  }

  float biasj[4], multj[4];
  #pragma unroll
  for (int j = 0; j < 4; ++j) {
    const int n = bn + wn + j * 16 + l16;
    biasj[j] = bias[n];
    multj[j] = (z < 2) ? attm[n] : 1.0f;   // att_mask folded into Q and K
  }
  #pragma unroll
  for (int i = 0; i < 2; ++i)
    #pragma unroll
    for (int j = 0; j < 4; ++j)
      #pragma unroll
      for (int r = 0; r < 8; ++r) {
        const int m  = bm + wm + i * 16 + r + lh * 8;   // C layout row
        const int n  = bn + wn + j * 16 + l16;          // C layout col
        const int bb = m >> 11, ss = m & (S_ - 1);
        const int hh = n >> 6,  dd = n & (DH_ - 1);
        const float v = (c[i][j][r] + biasj[j]) * multj[j];
        qkv[((((size_t)z * B_ + bb) * NH_ + hh) * S_ + ss) * DH_ + dd] = (__bf16)v;
      }
}

// ---------------------------------------------------------------------------
// Flash attention: one block (128 threads = 4 waves) per (64 q-rows, b, h).
// Each wave owns 16 q-rows. Online softmax, f32 stats, bf16 WMMA matmuls.
// Q/K tiles: async DMA to LDS (ASYNCcnt path, frees VALU for softmax).
// V tile: VGPR path (needs transpose on the way into LDS).
// ---------------------------------------------------------------------------
#define LQ 72    // padded LDS row stride (halves): 36 dwords, conflict-free

__global__ __launch_bounds__(128)
void attention(const __bf16* __restrict__ qkv, const float* __restrict__ amask,
               const float* __restrict__ attm, const float* __restrict__ valm,
               float* __restrict__ out)
{
  __shared__ __bf16 Qs[64][LQ];
  __shared__ __bf16 Ks[64][LQ];
  __shared__ __bf16 Vt[64][LQ];      // V transposed: Vt[d][key]
  __shared__ __bf16 Ps[4][16][LQ];   // per-wave P staging (C-layout -> A-layout)
  __shared__ float  s_scale;

  const int qt  = blockIdx.x;             // q tile (0..31)
  const int bh  = blockIdx.y;             // 0..31
  const int b   = bh >> 4, h = bh & 15;
  const int tid = threadIdx.x;
  const int lane = tid & 31, w = tid >> 5;
  const int l16  = lane & 15, lh = lane >> 4;

  const __bf16* Qg = qkv + (((size_t)0 * B_ + b) * NH_ + h) * (size_t)S_ * DH_
                         + (size_t)qt * 64 * DH_;
  const __bf16* Kg = qkv + (((size_t)1 * B_ + b) * NH_ + h) * (size_t)S_ * DH_;
  const __bf16* Vg = qkv + (((size_t)2 * B_ + b) * NH_ + h) * (size_t)S_ * DH_;

  if (tid == 0) {   // head_size = sum(att_mask[0,:]) -> softmax scale
    float s = 0.f;
    for (int i = 0; i < DH_; ++i) s += attm[i];
    s_scale = rsqrtf(s);
  }

  const int trow = tid >> 1;
  const int tcol = (tid & 1) * 32;
  const unsigned tile_voff = (unsigned)((trow * DH_ + tcol) * 2);  // bytes in 64x64 tile
  const unsigned ldsQ = lds_off(&Qs[trow][tcol]);
  const unsigned ldsK = lds_off(&Ks[trow][tcol]);

  // stage Q tile (64x64) via async DMA, 64B per thread
  async_copy64(ldsQ, (const void*)Qg, tile_voff);
  wait_async0();
  __syncthreads();
  const float scale = s_scale;

  Frag16 aq[2];   // Q A-fragments for the two 32-wide K-steps over DH
  #pragma unroll
  for (int ks = 0; ks < 2; ++ks) {
    const int m = w * 16 + l16;
    aq[ks].u[0] = *(const uint4*)&Qs[m][ks * 32 + lh * 8];
    aq[ks].u[1] = *(const uint4*)&Qs[m][ks * 32 + 16 + lh * 8];
  }

  v8f o[4];
  float mrow[8], lrow[8];
  #pragma unroll
  for (int r = 0; r < 8; ++r) { mrow[r] = -3.0e38f; lrow[r] = 0.0f; }
  #pragma unroll
  for (int j = 0; j < 4; ++j)
    for (int r = 0; r < 8; ++r) o[j][r] = 0.0f;

  for (int kt = 0; kt < S_ / 64; ++kt) {
    __syncthreads();   // previous iteration's LDS consumers are done

    if (kt + 1 < S_ / 64) {  // gfx1250 global_prefetch_b8 for next V tile
      __builtin_prefetch((const void*)(Vg + (size_t)(kt + 1) * 64 * DH_ + (size_t)trow * DH_ + tcol), 0, 0);
    }
    // stage K tile (64x64) row-major via async DMA (ASYNCcnt)
    async_copy64(ldsK, (const void*)(Kg + (size_t)kt * 64 * DH_), tile_voff);
    { // stage V tile transposed: Vt[d][key] (VGPR path)
      const uint4* g = (const uint4*)(Vg + (size_t)(kt * 64 + trow) * DH_ + tcol);
      union { uint4 u[4]; __bf16 hh[32]; } p;
      p.u[0] = g[0]; p.u[1] = g[1]; p.u[2] = g[2]; p.u[3] = g[3];
      #pragma unroll
      for (int i = 0; i < 32; ++i) Vt[tcol + i][trow] = p.hh[i];
    }
    wait_async0();
    __syncthreads();

    // ---- S = Q * K^T  (16 q-rows x 64 keys per wave) ----
    v8f sc[4];
    #pragma unroll
    for (int j = 0; j < 4; ++j)
      for (int r = 0; r < 8; ++r) sc[j][r] = 0.0f;
    #pragma unroll
    for (int ks = 0; ks < 2; ++ks)
      #pragma unroll
      for (int j = 0; j < 4; ++j) {
        Frag16 bk_;   // B = K^T: lane = key column, d split per lane-half
        bk_.u[0] = *(const uint4*)&Ks[j * 16 + l16][ks * 32 + lh * 8];
        bk_.u[1] = *(const uint4*)&Ks[j * 16 + l16][ks * 32 + 16 + lh * 8];
        sc[j] = __builtin_amdgcn_wmma_f32_16x16x32_bf16(
            false, aq[ks].v, false, bk_.v, (short)0, sc[j], false, false);
      }

    float am4[4];
    #pragma unroll
    for (int j = 0; j < 4; ++j)
      am4[j] = amask[(size_t)b * S_ + kt * 64 + j * 16 + l16];

    #pragma unroll
    for (int j = 0; j < 4; ++j)
      #pragma unroll
      for (int r = 0; r < 8; ++r)
        sc[j][r] = sc[j][r] * scale + am4[j];

    // ---- online softmax (row stats across the 16-lane group) ----
    #pragma unroll
    for (int r = 0; r < 8; ++r) {
      float mx = fmaxf(fmaxf(sc[0][r], sc[1][r]), fmaxf(sc[2][r], sc[3][r]));
      mx = fmaxf(mx, __shfl_xor(mx, 1));
      mx = fmaxf(mx, __shfl_xor(mx, 2));
      mx = fmaxf(mx, __shfl_xor(mx, 4));
      mx = fmaxf(mx, __shfl_xor(mx, 8));
      const float mnew  = fmaxf(mrow[r], mx);
      const float alpha = __expf(mrow[r] - mnew);
      mrow[r] = mnew;
      lrow[r] *= alpha;
      #pragma unroll
      for (int j = 0; j < 4; ++j) o[j][r] *= alpha;
      float rs = 0.f;
      #pragma unroll
      for (int j = 0; j < 4; ++j) {
        const float p = __expf(sc[j][r] - mnew);
        sc[j][r] = p;
        rs += p;
      }
      rs += __shfl_xor(rs, 1);
      rs += __shfl_xor(rs, 2);
      rs += __shfl_xor(rs, 4);
      rs += __shfl_xor(rs, 8);
      lrow[r] += rs;
    }

    // ---- reshape P: C-layout regs -> LDS -> A-layout fragments ----
    #pragma unroll
    for (int j = 0; j < 4; ++j)
      #pragma unroll
      for (int r = 0; r < 8; ++r)
        Ps[w][r + lh * 8][j * 16 + l16] = (__bf16)sc[j][r];
    __syncthreads();

    Frag16 ap[2];
    #pragma unroll
    for (int ks = 0; ks < 2; ++ks) {
      ap[ks].u[0] = *(const uint4*)&Ps[w][l16][ks * 32 + lh * 8];
      ap[ks].u[1] = *(const uint4*)&Ps[w][l16][ks * 32 + 16 + lh * 8];
    }
    // ---- O += P * V ----
    #pragma unroll
    for (int ks = 0; ks < 2; ++ks)
      #pragma unroll
      for (int j = 0; j < 4; ++j) {
        Frag16 bv_;   // B = V: lane = d column, key split per lane-half (from Vt)
        bv_.u[0] = *(const uint4*)&Vt[j * 16 + l16][ks * 32 + lh * 8];
        bv_.u[1] = *(const uint4*)&Vt[j * 16 + l16][ks * 32 + 16 + lh * 8];
        o[j] = __builtin_amdgcn_wmma_f32_16x16x32_bf16(
            false, ap[ks].v, false, bv_.v, (short)0, o[j], false, false);
      }
  }

  // ---- epilogue: normalize, val_mask, store f32 [b, s, h*64+d] ----
  #pragma unroll
  for (int r = 0; r < 8; ++r) {
    const float inv  = 1.0f / lrow[r];
    const int   srow = qt * 64 + w * 16 + r + lh * 8;
    #pragma unroll
    for (int j = 0; j < 4; ++j) {
      const int n = h * DH_ + j * 16 + l16;
      out[((size_t)b * S_ + srow) * (NH_ * DH_) + n] = o[j][r] * inv * valm[n];
    }
  }
}

// ---------------------------------------------------------------------------
// Host launcher
// ---------------------------------------------------------------------------
extern "C" void kernel_launch(void* const* d_in, const int* in_sizes, int n_in,
                              void* d_out, int out_size, void* d_ws, size_t ws_size,
                              hipStream_t stream) {
  (void)in_sizes; (void)n_in; (void)out_size; (void)ws_size;
  const float* hs    = (const float*)d_in[0];
  const float* amask = (const float*)d_in[1];
  const float* Wq    = (const float*)d_in[2];
  const float* bq    = (const float*)d_in[3];
  const float* Wk    = (const float*)d_in[4];
  const float* bk    = (const float*)d_in[5];
  const float* Wv    = (const float*)d_in[6];
  const float* bv    = (const float*)d_in[7];
  const float* attm  = (const float*)d_in[8];
  const float* valm  = (const float*)d_in[9];
  float* out = (float*)d_out;

  // Workspace layout: Xb (8 MiB) | Wb x3 (6 MiB) | QKV (24 MiB)
  char* ws = (char*)d_ws;
  __bf16* Xb  = (__bf16*)ws;
  __bf16* Wb  = (__bf16*)(ws + ((size_t)8  << 20));
  __bf16* QKV = (__bf16*)(ws + ((size_t)14 << 20));

  const int nX4 = (M_ * H_) / 4;   // 1,048,576
  const int nW4 = (H_ * H_) / 4;   //   262,144
  cast_f32_bf16<<<(nX4 + 255) / 256, 256, 0, stream>>>(hs, Xb, nX4);
  cast_f32_bf16<<<(nW4 + 255) / 256, 256, 0, stream>>>(Wq, Wb,               nW4);
  cast_f32_bf16<<<(nW4 + 255) / 256, 256, 0, stream>>>(Wk, Wb + H_ * H_,     nW4);
  cast_f32_bf16<<<(nW4 + 255) / 256, 256, 0, stream>>>(Wv, Wb + 2 * H_ * H_, nW4);

  qkv_gemm<<<dim3(H_ / BN, M_ / BM, 3), 256, 0, stream>>>(
      Xb, Wb, bq, bk, bv, attm, QKV);

  attention<<<dim3(S_ / 64, B_ * NH_), 128, 0, stream>>>(
      QKV, amask, attm, valm, out);
}
